// TransMIL_63333587747609
// MI455X (gfx1250) — matrix operand order repaired and checked by reading, not compiled, not run on hardware
//
#include <hip/hip_runtime.h>
#include <hip/hip_bf16.h>

// ---------------------------------------------------------------------------
// Types for CDNA5 WMMA (wave32)
// ---------------------------------------------------------------------------
typedef __attribute__((ext_vector_type(16))) __bf16 v16bf;
typedef __attribute__((ext_vector_type(8)))  float  v8f;

#define BATCH 4
#define NRAW  6000
#define NTOK  6085      // cls + 6084 grid tokens
#define NP    6144      // landmark-padded length
#define PADLO 59        // zero rows prepended
#define EMB   512
#define NH    8
#define DH    64
#define NLM   256
#define GRIDH 78

__device__ __forceinline__ unsigned short f32_to_bf16(float f) {
    unsigned int u = __float_as_uint(f);
    u += 0x7FFFu + ((u >> 16) & 1u);          // round-to-nearest-even
    return (unsigned short)(u >> 16);
}

// ---------------------------------------------------------------------------
// bf16 WMMA GEMM:  C = beta*C + alpha*(A @ B) (+bias) (ReLU)
// A: MxK f32 (lda), B: KxN f32 (ldb) [NN] or NxK f32 (ldb) [NT], C: MxN f32.
// Batched via blockIdx.z decomposed z = z0*nz1 + z1 with two-level strides.
//
// Block tile 128x128, K-step 32, 8 waves as 4(m) x 2(n); each wave owns a
// 32x64 strip = 2 A-frags x 4 B-frags -> 8 WMMA accumulators.
// LDS tiles are double-buffered and stored fragment-ready:
//   As [buf][row][k] pitch 40  -> A frag = two aligned 16B chunks per lane
//   BsT[buf][col][k] pitch 40  -> B frag = two aligned 16B chunks per lane
// so every fragment load is 2 x ds_load_b128 (pitch 40 ushorts = 80B keeps
// 16B alignment and walks all 64 banks across 16 consecutive rows).
// Double buffering: stage tile k+1 (global->LDS) while WMMAs consume tile k;
// one barrier per K-step.
// ---------------------------------------------------------------------------
#define BM 128
#define BN 128
#define BK 32
#define LPITCH 40

union Frag {
    uint4 u[2];
    v16bf v;
};

template <int NT_MODE>
__global__ __launch_bounds__(256)
void gemm_bf16_k(const float* __restrict__ A, const float* __restrict__ B,
                 float* __restrict__ C, const float* __restrict__ bias,
                 int M, int N, int K, int lda, int ldb, int ldc,
                 long long sA0, long long sA1, long long sB0, long long sB1,
                 long long sC0, long long sC1, int nz1,
                 float alpha, int beta, int relu)
{
    int z  = blockIdx.z;
    int z0 = z / nz1, z1 = z % nz1;
    A += z0 * sA0 + z1 * sA1;
    B += z0 * sB0 + z1 * sB1;
    C += z0 * sC0 + z1 * sC1;

    __shared__ __align__(16) unsigned short As [2][BM][LPITCH];
    __shared__ __align__(16) unsigned short BsT[2][BN][LPITCH];

    const int tid  = threadIdx.x;
    const int wave = tid >> 5;
    const int lane = tid & 31;
    const int half = lane >> 4;     // 0: lanes 0-15, 1: lanes 16-31
    const int l16  = lane & 15;
    const int wm   = wave >> 1;     // 0..3  (32-row strip)
    const int wn   = wave & 1;      // 0..1  (64-col strip)
    const int m0   = blockIdx.y * BM;
    const int n0   = blockIdx.x * BN;

    // ---- tile staging (global f32 -> bf16 LDS), fragment-ready layouts ----
    auto stage = [&](int buf, int kk) {
        {   // A tile (128x32): 16 consecutive floats/thread -> 2x b128 LDS store
            int r  = tid >> 1;             // 0..127
            int ck = (tid & 1) * 16;       // 0 or 16
            int gr = m0 + r; if (gr >= M) gr = M - 1;
            const float* ap = A + (long long)gr * lda + kk + ck;
            __builtin_prefetch(ap + BK, 0, 1);   // global_prefetch next K tile
            union { unsigned short s[16]; uint4 u[2]; } t;
#pragma unroll
            for (int j = 0; j < 16; ++j) t.s[j] = f32_to_bf16(ap[j]);
            *(uint4*)&As[buf][r][ck]     = t.u[0];
            *(uint4*)&As[buf][r][ck + 8] = t.u[1];
        }
        if (NT_MODE == 0) {
            // B is KxN: BsT[n][k] = B[k][n]; pack K-pairs -> aligned b32 stores
            int kp = (tid & 15) * 2;       // even k
            int ng = (tid >> 4) * 8;       // 8 columns per thread
            const float* bp0 = B + (long long)(kk + kp) * ldb;
            const float* bp1 = bp0 + ldb;
#pragma unroll
            for (int j = 0; j < 8; ++j) {
                int gc = n0 + ng + j; if (gc >= N) gc = N - 1;
                unsigned int lo = f32_to_bf16(bp0[gc]);
                unsigned int hi = f32_to_bf16(bp1[gc]);
                *(unsigned int*)&BsT[buf][ng + j][kp] = lo | (hi << 16);
            }
        } else {
            // B is NxK: straight copy, 16 consecutive K/thread -> 2x b128 store
            int r  = tid >> 1;             // n: 0..127
            int ck = (tid & 1) * 16;
            int gn = n0 + r; if (gn >= N) gn = N - 1;
            const float* bp = B + (long long)gn * ldb + kk + ck;
            __builtin_prefetch(bp + BK, 0, 1);
            union { unsigned short s[16]; uint4 u[2]; } t;
#pragma unroll
            for (int j = 0; j < 16; ++j) t.s[j] = f32_to_bf16(bp[j]);
            *(uint4*)&BsT[buf][r][ck]     = t.u[0];
            *(uint4*)&BsT[buf][r][ck + 8] = t.u[1];
        }
    };

    v8f acc[2][4];
#pragma unroll
    for (int am = 0; am < 2; ++am)
#pragma unroll
        for (int nb = 0; nb < 4; ++nb)
#pragma unroll
            for (int r = 0; r < 8; ++r) acc[am][nb][r] = 0.0f;

    stage(0, 0);
    __syncthreads();

    int cur = 0;
    for (int k0 = 0; k0 < K; k0 += BK) {
        // stage next tile into the other buffer while this one is consumed
        if (k0 + BK < K) stage(cur ^ 1, k0 + BK);

        // ---- A fragments: ISA 16-bit A layout = two 16B runs per lane ----
        Frag af[2];
#pragma unroll
        for (int am = 0; am < 2; ++am) {
            const int arow = wm * 32 + am * 16 + l16;
            af[am].u[0] = *(const uint4*)&As[cur][arow][half * 8];
            af[am].u[1] = *(const uint4*)&As[cur][arow][16 + half * 8];
        }
        // ---- B fragments + 8 WMMAs ----
#pragma unroll
        for (int nb = 0; nb < 4; ++nb) {
            Frag bf;
            const int bcol = wn * 64 + nb * 16 + l16;
            bf.u[0] = *(const uint4*)&BsT[cur][bcol][half * 8];
            bf.u[1] = *(const uint4*)&BsT[cur][bcol][16 + half * 8];
#pragma unroll
            for (int am = 0; am < 2; ++am) {
                acc[am][nb] = __builtin_amdgcn_wmma_f32_16x16x32_bf16(
                    false, af[am].v, false, bf.v, (short)0, acc[am][nb], false, false);
            }
        }
        __syncthreads();   // next buffer fully staged; current fully consumed
        cur ^= 1;
    }

    // ---- epilogue: D layout VGPR r -> M = r + 8*half, N = l16 ----
#pragma unroll
    for (int am = 0; am < 2; ++am) {
#pragma unroll
        for (int nb = 0; nb < 4; ++nb) {
            int gn = n0 + wn * 64 + nb * 16 + l16;
#pragma unroll
            for (int r = 0; r < 8; ++r) {
                int gm = m0 + wm * 32 + am * 16 + half * 8 + r;
                if (gm < M && gn < N) {
                    float v = alpha * acc[am][nb][r];
                    if (bias) v += bias[gn];
                    if (beta) v += C[(long long)gm * ldc + gn];
                    if (relu) v = fmaxf(v, 0.0f);
                    C[(long long)gm * ldc + gn] = v;
                }
            }
        }
    }
}

// ---------------------------------------------------------------------------
// Row softmax, one 256-thread block per row of length L (in place).
// ---------------------------------------------------------------------------
__global__ __launch_bounds__(256)
void softmax_rows_k(float* __restrict__ X, int L)
{
    float* x = X + (long long)blockIdx.x * L;
    __shared__ float red[256];
    int tid = threadIdx.x;
    float m = -3.0e38f;
    for (int i = tid; i < L; i += 256) m = fmaxf(m, x[i]);
    red[tid] = m; __syncthreads();
    for (int s = 128; s > 0; s >>= 1) { if (tid < s) red[tid] = fmaxf(red[tid], red[tid + s]); __syncthreads(); }
    m = red[0]; __syncthreads();
    float sum = 0.0f;
    for (int i = tid; i < L; i += 256) { float e = __expf(x[i] - m); x[i] = e; sum += e; }
    red[tid] = sum; __syncthreads();
    for (int s = 128; s > 0; s >>= 1) { if (tid < s) red[tid] += red[tid + s]; __syncthreads(); }
    float inv = 1.0f / red[0];
    for (int i = tid; i < L; i += 256) x[i] *= inv;
}

// ---------------------------------------------------------------------------
// LayerNorm(tok) -> zero-padded x_p (rows [0,PADLO) = 0, else LN(tok[r-PADLO]))
// grid (NP, BATCH)
// ---------------------------------------------------------------------------
__global__ __launch_bounds__(256)
void ln_pad_k(const float* __restrict__ tok, const float* __restrict__ g,
              const float* __restrict__ b, float* __restrict__ xp)
{
    int row = blockIdx.x, bi = blockIdx.y, tid = threadIdx.x;
    float* out = xp + ((long long)bi * NP + row) * EMB;
    if (row < PADLO) { for (int i = tid; i < EMB; i += 256) out[i] = 0.0f; return; }
    const float* in = tok + ((long long)bi * NTOK + (row - PADLO)) * EMB;
    __shared__ float red[256];
    float s = 0.0f;
    for (int i = tid; i < EMB; i += 256) s += in[i];
    red[tid] = s; __syncthreads();
    for (int st = 128; st > 0; st >>= 1) { if (tid < st) red[tid] += red[tid + st]; __syncthreads(); }
    float mu = red[0] * (1.0f / EMB); __syncthreads();
    float v = 0.0f;
    for (int i = tid; i < EMB; i += 256) { float d = in[i] - mu; v += d * d; }
    red[tid] = v; __syncthreads();
    for (int st = 128; st > 0; st >>= 1) { if (tid < st) red[tid] += red[tid + st]; __syncthreads(); }
    float inv = rsqrtf(red[0] * (1.0f / EMB) + 1e-5f);
    for (int i = tid; i < EMB; i += 256) out[i] = (in[i] - mu) * inv * g[i] + b[i];
}

// cls row + wrap-around pad rows.  grid (85, BATCH)
__global__ void build_tokens_k(float* __restrict__ tok, const float* __restrict__ cls)
{
    int r = blockIdx.x, b = blockIdx.y;
    float* dst; const float* src;
    if (r == 0) { dst = tok + (long long)b * NTOK * EMB; src = cls; }
    else        { dst = tok + ((long long)b * NTOK + NRAW + r) * EMB;
                  src = tok + ((long long)b * NTOK + r) * EMB; }
    for (int c = threadIdx.x; c < EMB; c += 256) dst[c] = src[c];
}

// landmark means: ql (scaled by 1/8), kl.  grid (NLM, 32), block DH
__global__ void landmarks_k(const float* __restrict__ qkv,
                            float* __restrict__ ql, float* __restrict__ kl)
{
    int j = blockIdx.x, bh = blockIdx.y, d = threadIdx.x;
    int b = bh >> 3, h = bh & 7;
    const float* base = qkv + (long long)b * NP * (3 * EMB) + h * DH + d;
    float sq = 0.0f, sk = 0.0f;
#pragma unroll 4
    for (int i = 0; i < NP / NLM; ++i) {
        long long t = (long long)(j * (NP / NLM) + i) * (3 * EMB);
        sq += base[t];
        sk += base[t + EMB];
    }
    long long o = ((long long)bh * NLM + j) * DH + d;
    const float il = 1.0f / (float)(NP / NLM);
    ql[o] = sq * 0.125f * il;
    kl[o] = sk * il;
}

// ------------------- pinv helpers -------------------
__global__ void zero_k(float* p, int n)
{ int i = blockIdx.x * 256 + threadIdx.x; if (i < n) p[i] = 0.0f; }

__global__ __launch_bounds__(256)
void abs_rowsum_max_k(const float* __restrict__ x, float* __restrict__ gmax)
{
    const float* r = x + ((long long)blockIdx.y * NLM + blockIdx.x) * NLM;
    __shared__ float red[256];
    red[threadIdx.x] = fabsf(r[threadIdx.x]); __syncthreads();
    for (int st = 128; st > 0; st >>= 1) { if (threadIdx.x < st) red[threadIdx.x] += red[threadIdx.x + st]; __syncthreads(); }
    if (threadIdx.x == 0) atomicMax((int*)gmax, __float_as_int(red[0]));
}
__global__ __launch_bounds__(256)
void abs_colsum_max_k(const float* __restrict__ x, float* __restrict__ gmax)
{
    const float* m = x + (long long)blockIdx.x * NLM * NLM;
    float s = 0.0f;
    for (int i = 0; i < NLM; ++i) s += fabsf(m[(long long)i * NLM + threadIdx.x]);
    atomicMax((int*)gmax, __float_as_int(s));
}
__global__ void zinit_k(const float* __restrict__ x, float* __restrict__ z,
                        const float* __restrict__ cmax, const float* __restrict__ rmax)
{
    long long bh = blockIdx.y; int i = blockIdx.x, j = threadIdx.x;
    float inv = 1.0f / (cmax[0] * rmax[0]);
    z[bh * NLM * NLM + (long long)i * NLM + j] =
        x[bh * NLM * NLM + (long long)j * NLM + i] * inv;
}
__global__ void subI_k(const float* __restrict__ in, float* __restrict__ out, float c)
{
    long long bh = blockIdx.y; int i = blockIdx.x, j = threadIdx.x;
    long long o = bh * NLM * NLM + (long long)i * NLM + j;
    out[o] = (i == j ? c : 0.0f) - in[o];
}

// depthwise 33-tap sequence conv residual: attn[b,t,h*64+d] += sum_j w[h,j]*v[b,t+j-16,h,d]
__global__ void seqconv_add_k(const float* __restrict__ qkv, const float* __restrict__ w,
                              float* __restrict__ attn)
{
    long long idx = (long long)blockIdx.x * 256 + threadIdx.x;
    if (idx >= (long long)BATCH * NP * EMB) return;
    int c = idx & (EMB - 1); long long bt = idx >> 9;
    int t = (int)(bt % NP); int b = (int)(bt / NP);
    int h = c >> 6, d = c & 63;
    const float* vb = qkv + (long long)b * NP * (3 * EMB) + 2 * EMB + h * DH + d;
    float s = 0.0f;
#pragma unroll
    for (int j = 0; j < 33; ++j) {
        int tt = t + j - 16;
        if (tt >= 0 && tt < NP) s += w[h * 33 + j] * vb[(long long)tt * (3 * EMB)];
    }
    attn[idx] += s;
}

// PPEG: tokout[b,1+p,c] = f + dw7 + b7 + dw5 + b5 + dw3 + b3 over 78x78 grid
__global__ void ppeg_k(const float* __restrict__ tin, float* __restrict__ tout,
                       const float* __restrict__ w7, const float* __restrict__ b7,
                       const float* __restrict__ w5, const float* __restrict__ b5,
                       const float* __restrict__ w3, const float* __restrict__ b3)
{
    long long idx = (long long)blockIdx.x * 256 + threadIdx.x;
    long long total = (long long)BATCH * EMB * (GRIDH * GRIDH);
    if (idx >= total) return;
    int p = (int)(idx % (GRIDH * GRIDH)); long long bc = idx / (GRIDH * GRIDH);
    int c = (int)(bc % EMB); int b = (int)(bc / EMB);
    int i = p / GRIDH, j = p % GRIDH;
    const float* f = tin + ((long long)b * NTOK + 1) * EMB + c;
    float acc = f[(long long)p * EMB];
    float s7 = 0.0f, s5 = 0.0f, s3 = 0.0f;
    for (int dy = -3; dy <= 3; ++dy)
        for (int dx = -3; dx <= 3; ++dx) {
            int y = i + dy, x = j + dx;
            if (y < 0 || y >= GRIDH || x < 0 || x >= GRIDH) continue;
            float fv = f[(long long)(y * GRIDH + x) * EMB];
            s7 += w7[c * 49 + (dy + 3) * 7 + (dx + 3)] * fv;
            if (dy >= -2 && dy <= 2 && dx >= -2 && dx <= 2)
                s5 += w5[c * 25 + (dy + 2) * 5 + (dx + 2)] * fv;
            if (dy >= -1 && dy <= 1 && dx >= -1 && dx <= 1)
                s3 += w3[c * 9 + (dy + 1) * 3 + (dx + 1)] * fv;
        }
    acc += s7 + b7[c] + s5 + b5[c] + s3 + b3[c];
    tout[((long long)b * NTOK + 1 + p) * EMB + c] = acc;
}
__global__ void copy_cls_k(const float* __restrict__ tin, float* __restrict__ tout)
{
    int b = blockIdx.x;
    for (int c = threadIdx.x; c < EMB; c += 256)
        tout[(long long)b * NTOK * EMB + c] = tin[(long long)b * NTOK * EMB + c];
}

// final: LN(cls) -> fc2 -> hazards/S/Y_hat.  grid BATCH
__global__ __launch_bounds__(256)
void head_k(const float* __restrict__ tok, const float* __restrict__ g,
            const float* __restrict__ bt, const float* __restrict__ w,
            const float* __restrict__ bb, float* __restrict__ out)
{
    int b = blockIdx.x, tid = threadIdx.x;
    const float* x = tok + (long long)b * NTOK * EMB;
    __shared__ float red[256];
    __shared__ float lg[4];
    float s = 0.0f;
    for (int i = tid; i < EMB; i += 256) s += x[i];
    red[tid] = s; __syncthreads();
    for (int st = 128; st > 0; st >>= 1) { if (tid < st) red[tid] += red[tid + st]; __syncthreads(); }
    float mu = red[0] * (1.0f / EMB); __syncthreads();
    float v = 0.0f;
    for (int i = tid; i < EMB; i += 256) { float d = x[i] - mu; v += d * d; }
    red[tid] = v; __syncthreads();
    for (int st = 128; st > 0; st >>= 1) { if (tid < st) red[tid] += red[tid + st]; __syncthreads(); }
    float inv = rsqrtf(red[0] * (1.0f / EMB) + 1e-5f); __syncthreads();
    for (int k = 0; k < 4; ++k) {
        float p = 0.0f;
        for (int i = tid; i < EMB; i += 256) {
            float xn = (x[i] - mu) * inv * g[i] + bt[i];
            p += xn * w[i * 4 + k];
        }
        red[tid] = p; __syncthreads();
        for (int st = 128; st > 0; st >>= 1) { if (tid < st) red[tid] += red[tid + st]; __syncthreads(); }
        if (tid == 0) lg[k] = red[0] + bb[k];
        __syncthreads();
    }
    if (tid == 0) {
        int best = 0;
        for (int k = 1; k < 4; ++k) if (lg[k] > lg[best]) best = k;
        float S = 1.0f;
        for (int k = 0; k < 4; ++k) {
            float hz = 1.0f / (1.0f + __expf(-lg[k]));
            out[b * 4 + k]      = hz;      // hazards (4,4)
            S *= (1.0f - hz);
            out[16 + b * 4 + k] = S;       // survival (4,4)
        }
        out[32 + b] = (float)best;          // Y_hat (4,)
    }
}

// ---------------------------------------------------------------------------
// Host orchestration
// ---------------------------------------------------------------------------
extern "C" void kernel_launch(void* const* d_in, const int* in_sizes, int n_in,
                              void* d_out, int out_size, void* d_ws, size_t ws_size,
                              hipStream_t stream)
{
    (void)in_sizes; (void)n_in; (void)out_size; (void)ws_size;
    const float* x_path = (const float*)d_in[0];
    const float* fc1_w  = (const float*)d_in[1];
    const float* fc1_b  = (const float*)d_in[2];
    const float* cls_t  = (const float*)d_in[3];
    const float* ln1_g  = (const float*)d_in[4];
    const float* ln1_b  = (const float*)d_in[5];
    const float* qkv1_w = (const float*)d_in[6];
    const float* out1_w = (const float*)d_in[7];
    const float* out1_b = (const float*)d_in[8];
    const float* res1_w = (const float*)d_in[9];
    const float* ln2_g  = (const float*)d_in[10];
    const float* ln2_b  = (const float*)d_in[11];
    const float* qkv2_w = (const float*)d_in[12];
    const float* out2_w = (const float*)d_in[13];
    const float* out2_b = (const float*)d_in[14];
    const float* res2_w = (const float*)d_in[15];
    const float* c7w = (const float*)d_in[16]; const float* c7b = (const float*)d_in[17];
    const float* c5w = (const float*)d_in[18]; const float* c5b = (const float*)d_in[19];
    const float* c3w = (const float*)d_in[20]; const float* c3b = (const float*)d_in[21];
    const float* nrm_g = (const float*)d_in[22];
    const float* nrm_b = (const float*)d_in[23];
    const float* fc2_w = (const float*)d_in[24];
    const float* fc2_b = (const float*)d_in[25];

    // ---- workspace carve-up ----
    char* ws = (char*)d_ws; size_t off = 0;
    auto carve = [&](size_t bytes) -> float* {
        float* p = (float*)(ws + off);
        off = (off + bytes + 255) & ~(size_t)255;
        return p;
    };
    float* htok  = carve((size_t)BATCH * NTOK * EMB * 4);
    float* htok2 = carve((size_t)BATCH * NTOK * EMB * 4);
    float* xp    = carve((size_t)BATCH * NP * EMB * 4);
    float* qkvb  = carve((size_t)BATCH * NP * 3 * EMB * 4);
    float* ql    = carve((size_t)32 * NLM * DH * 4);
    float* kl    = carve((size_t)32 * NLM * DH * 4);
    float* sbuf  = carve((size_t)32 * NLM * NP * 4);   // shared by a3 then a1
    float* a2b   = carve((size_t)32 * NLM * NLM * 4);
    float* zb    = carve((size_t)32 * NLM * NLM * 4);
    float* z2b   = carve((size_t)32 * NLM * NLM * 4);
    float* xzb   = carve((size_t)32 * NLM * NLM * 4);
    float* tAb   = carve((size_t)32 * NLM * NLM * 4);
    float* tBb   = carve((size_t)32 * NLM * NLM * 4);
    float* avb   = carve((size_t)32 * NLM * DH * 4);
    float* zvb   = carve((size_t)32 * NLM * DH * 4);
    float* attn  = carve((size_t)BATCH * NP * EMB * 4);
    float* scal  = carve(256);

    const dim3 blk(256);
    auto gemm_nn = [&](const float* A, const float* B, float* C, const float* bias,
                       int M, int N, int K, int lda, int ldb, int ldc,
                       long long sA0, long long sA1, long long sB0, long long sB1,
                       long long sC0, long long sC1, int nz1, int nz,
                       float alpha, int beta, int relu) {
        dim3 g((N + BN - 1) / BN, (M + BM - 1) / BM, nz);
        gemm_bf16_k<0><<<g, blk, 0, stream>>>(A, B, C, bias, M, N, K, lda, ldb, ldc,
                                              sA0, sA1, sB0, sB1, sC0, sC1, nz1, alpha, beta, relu);
    };
    auto gemm_nt = [&](const float* A, const float* B, float* C, const float* bias,
                       int M, int N, int K, int lda, int ldb, int ldc,
                       long long sA0, long long sA1, long long sB0, long long sB1,
                       long long sC0, long long sC1, int nz1, int nz,
                       float alpha, int beta, int relu) {
        dim3 g((N + BN - 1) / BN, (M + BM - 1) / BM, nz);
        gemm_bf16_k<1><<<g, blk, 0, stream>>>(A, B, C, bias, M, N, K, lda, ldb, ldc,
                                              sA0, sA1, sB0, sB1, sC0, sC1, nz1, alpha, beta, relu);
    };
    auto softmax = [&](float* X, long long rows, int L) {
        softmax_rows_k<<<dim3((unsigned)rows), blk, 0, stream>>>(X, L);
    };

    const long long sQKVb = (long long)NP * 3 * EMB;   // qkv batch stride (per b)
    const long long sBH   = (long long)NLM * DH;       // 16384

    auto attention = [&](float* tok, const float* lng, const float* lnb,
                         const float* qkvw, const float* outw, const float* outb,
                         const float* resw) {
        // layernorm + zero-pad
        ln_pad_k<<<dim3(NP, BATCH), blk, 0, stream>>>(tok, lng, lnb, xp);
        // qkv = x_p @ qkv_w          (batched over b)
        gemm_nn(xp, qkvw, qkvb, nullptr, NP, 3 * EMB, EMB, EMB, 3 * EMB, 3 * EMB,
                (long long)NP * EMB, 0, 0, 0, sQKVb, 0, 1, BATCH, 1.0f, 0, 0);
        // landmarks
        landmarks_k<<<dim3(NLM, 32), dim3(DH), 0, stream>>>(qkvb, ql, kl);
        // a2 = softmax(q_l @ k_l^T)
        gemm_nt(ql, kl, a2b, nullptr, NLM, NLM, DH, DH, DH, NLM,
                sBH, 0, sBH, 0, (long long)NLM * NLM, 0, 1, 32, 1.0f, 0, 0);
        softmax(a2b, 32LL * NLM, NLM);
        // pinv init
        zero_k<<<dim3(1), dim3(64), 0, stream>>>(scal, 2);
        abs_rowsum_max_k<<<dim3(NLM, 32), blk, 0, stream>>>(a2b, scal + 0);
        abs_colsum_max_k<<<dim3(32), blk, 0, stream>>>(a2b, scal + 1);
        zinit_k<<<dim3(NLM, 32), blk, 0, stream>>>(a2b, zb, scal + 0, scal + 1);
        float* zc = zb; float* zn = z2b;
        const long long sMM = (long long)NLM * NLM;
        for (int it = 0; it < 6; ++it) {
            gemm_nn(a2b, zc, xzb, nullptr, NLM, NLM, NLM, NLM, NLM, NLM,
                    sMM, 0, sMM, 0, sMM, 0, 1, 32, 1.0f, 0, 0);
            subI_k<<<dim3(NLM, 32), blk, 0, stream>>>(xzb, tAb, 7.0f);
            gemm_nn(xzb, tAb, tBb, nullptr, NLM, NLM, NLM, NLM, NLM, NLM,
                    sMM, 0, sMM, 0, sMM, 0, 1, 32, 1.0f, 0, 0);
            subI_k<<<dim3(NLM, 32), blk, 0, stream>>>(tBb, tAb, 15.0f);
            gemm_nn(xzb, tAb, tBb, nullptr, NLM, NLM, NLM, NLM, NLM, NLM,
                    sMM, 0, sMM, 0, sMM, 0, 1, 32, 1.0f, 0, 0);
            subI_k<<<dim3(NLM, 32), blk, 0, stream>>>(tBb, tAb, 13.0f);
            gemm_nn(zc, tAb, zn, nullptr, NLM, NLM, NLM, NLM, NLM, NLM,
                    sMM, 0, sMM, 0, sMM, 0, 1, 32, 0.25f, 0, 0);
            float* t = zc; zc = zn; zn = t;
        }
        // a3 = softmax(q_l @ k^T)  -> sbuf  (k rows live in qkv with pitch 1536)
        gemm_nt(ql, qkvb + EMB, sbuf, nullptr, NLM, NP, DH, DH, 3 * EMB, NP,
                8 * sBH, sBH, sQKVb, DH, 8LL * NLM * NP, (long long)NLM * NP,
                8, 32, 1.0f, 0, 0);
        softmax(sbuf, 32LL * NLM, NP);
        // av = a3 @ v
        gemm_nn(sbuf, qkvb + 2 * EMB, avb, nullptr, NLM, DH, NP, NP, 3 * EMB, DH,
                8LL * NLM * NP, (long long)NLM * NP, sQKVb, DH,
                8 * sBH, sBH, 8, 32, 1.0f, 0, 0);
        // zv = pinv(a2) @ av
        gemm_nn(zc, avb, zvb, nullptr, NLM, DH, NLM, NLM, DH, DH,
                sMM, 0, sBH, 0, sBH, 0, 1, 32, 1.0f, 0, 0);
        // a1 = softmax( (q/8) @ k_l^T )  -> sbuf (reuse)
        gemm_nt(qkvb, kl, sbuf, nullptr, NP, NLM, DH, 3 * EMB, DH, NLM,
                sQKVb, DH, 8 * sBH, sBH, 8LL * NP * NLM, (long long)NP * NLM,
                8, 32, 0.125f, 0, 0);
        softmax(sbuf, 32LL * NP, NLM);
        // out_heads = a1 @ zv  -> attn laid out (b, token, h*64+d)
        gemm_nn(sbuf, zvb, attn, nullptr, NP, DH, NLM, NLM, DH, EMB,
                8LL * NP * NLM, (long long)NP * NLM, 8 * sBH, sBH,
                (long long)NP * EMB, DH, 8, 32, 1.0f, 0, 0);
        // += depthwise 33-tap conv of v
        long long tot = (long long)BATCH * NP * EMB;
        seqconv_add_k<<<dim3((unsigned)((tot + 255) / 256)), blk, 0, stream>>>(qkvb, resw, attn);
        // tok += proj(attn rows [PADLO, NP)) + bias     (residual, beta=1)
        gemm_nn(attn + PADLO * EMB, outw, tok, outb, NTOK, EMB, EMB, EMB, EMB, EMB,
                (long long)NP * EMB, 0, 0, 0, (long long)NTOK * EMB, 0,
                1, BATCH, 1.0f, 1, 0);
    };

    // ---- pipeline ----
    // fc1 + relu -> token rows 1..6000
    gemm_nn(x_path, fc1_w, htok + EMB, fc1_b, NRAW, EMB, 1024, 1024, EMB, EMB,
            (long long)NRAW * 1024, 0, 0, 0, (long long)NTOK * EMB, 0,
            1, BATCH, 1.0f, 0, 1);
    build_tokens_k<<<dim3(85, BATCH), blk, 0, stream>>>(htok, cls_t);

    attention(htok, ln1_g, ln1_b, qkv1_w, out1_w, out1_b, res1_w);

    copy_cls_k<<<dim3(BATCH), blk, 0, stream>>>(htok, htok2);
    {
        long long tot = (long long)BATCH * EMB * (GRIDH * GRIDH);
        ppeg_k<<<dim3((unsigned)((tot + 255) / 256)), blk, 0, stream>>>(
            htok, htok2, c7w, c7b, c5w, c5b, c3w, c3b);
    }

    attention(htok2, ln2_g, ln2_b, qkv2_w, out2_w, out2_b, res2_w);

    head_k<<<dim3(BATCH), blk, 0, stream>>>(htok2, nrm_g, nrm_b, fc2_w, fc2_b, (float*)d_out);
}
